// GraphConvLayer_56049323213106
// MI455X (gfx1250) — matrix operand order repaired and checked by reading, not compiled, not run on hardware
//
#include <hip/hip_runtime.h>
#include <hip/hip_bf16.h>

#define N_NODES 50000
#define N_EDGES 800000
#define FEAT    64
#define IN_CH   128
#define OUT_CH  64

typedef __attribute__((ext_vector_type(8)))  _Float16 v8h;
typedef __attribute__((ext_vector_type(16))) _Float16 v16h;
typedef __attribute__((ext_vector_type(8)))  float    v8f;
typedef __attribute__((ext_vector_type(4)))  float    v4f;
typedef __attribute__((ext_vector_type(4)))  _Float16 v4h;

__device__ __forceinline__ v16h pack16(v8h lo, v8h hi) {
    v16h r;
#pragma unroll
    for (int i = 0; i < 8; ++i) { r[i] = lo[i]; r[8 + i] = hi[i]; }
    return r;
}

// ---------------------------------------------------------------------------
// Pass 1: f32 -> f16 conversion, 4 elements per thread (b128 in, b64 out)
// ---------------------------------------------------------------------------
__global__ __launch_bounds__(256) void cvt_f32_f16_x4(const float* __restrict__ in,
                                                      _Float16* __restrict__ out, int n4) {
    int i = blockIdx.x * blockDim.x + threadIdx.x;
    if (i < n4) {
        v4f f = *(const v4f*)(in + (size_t)i * 4);
        v4h h;
#pragma unroll
        for (int k = 0; k < 4; ++k) h[k] = (_Float16)f[k];
        *(v4h*)(out + (size_t)i * 4) = h;
    }
}

// ---------------------------------------------------------------------------
// Pass 2: per-edge GEMM via v_wmma_f32_16x16x32_f16.
// One wave32 owns a 16-edge x 64-output tile. Four independent accumulator
// chains (one per 16-wide N-tile) interleaved over the K loop so dependent
// WMMAs are >=4 apart -> RAW hazard slots co-execute useful work, no v_nops.
// ---------------------------------------------------------------------------
__global__ __launch_bounds__(256) void edge_gemm(
    const _Float16* __restrict__ xh,   // N_NODES x 64, f16 row-major
    const _Float16* __restrict__ wh,   // 64 x 128,     f16 row-major
    const long long* __restrict__ ei,  // 2 x N_EDGES int64
    const float* __restrict__ bias,    // 64
    float* __restrict__ y,             // N_EDGES x 64 output
    float* __restrict__ col0)          // N_EDGES snapshot of column 0
{
    const int lane = threadIdx.x & 31;
    const int tile = blockIdx.x * (blockDim.x >> 5) + (threadIdx.x >> 5);
    if (tile >= N_EDGES / 16) return;          // uniform per wave: EXEC stays all-1s

    const int m  = lane & 15;                  // edge-in-tile (A row / B,D column)
    const int hi = lane >> 4;                  // half-wave selector
    const int e0 = tile * 16;
    const int e  = e0 + m;

    const int src = (int)ei[e];
    const int dst = (int)ei[N_EDGES + e];

    // ---- Gather A: 4 K-chunks of 16x32 f16 in WMMA A-layout ----
    // lane m   : row m, local K {0..7, 16..23}
    // lane m+16: row m, local K {8..15, 24..31}
    v16h A[4];
#pragma unroll
    for (int kc = 0; kc < 4; ++kc) {
        const int node = (kc < 2) ? src : dst;
        const _Float16* p = xh + (size_t)node * FEAT + (kc & 1) * 32 + hi * 8;
        A[kc] = pack16(*(const v8h*)p, *(const v8h*)(p + 16));
    }

    // ---- hoist bias (column index == m for both half-waves) ----
    float bv[4];
#pragma unroll
    for (int nt = 0; nt < 4; ++nt) bv[nt] = bias[nt * 16 + m];

    // ---- 4 independent accumulator chains, K-loop outer ----
    v8f c[4];
#pragma unroll
    for (int nt = 0; nt < 4; ++nt) c[nt] = (v8f){};

#pragma unroll
    for (int kc = 0; kc < 4; ++kc) {
#pragma unroll
        for (int nt = 0; nt < 4; ++nt) {
            // B layout (32x16): lane n holds W[nt*16+n, kb+{0..7,16..23}],
            //                   lane n+16 holds W[nt*16+n, kb+{8..15,24..31}]
            const _Float16* p = wh + (size_t)(nt * 16 + m) * IN_CH + kc * 32 + hi * 8;
            const v16h b = pack16(*(const v8h*)p, *(const v8h*)(p + 16));
            c[nt] = __builtin_amdgcn_wmma_f32_16x16x32_f16(
                        /*neg_a=*/false, A[kc], /*neg_b=*/false, b,
                        /*c_mod=*/(short)0, c[nt], /*reuse_a=*/false, /*reuse_b=*/false);
        }
    }

    // ---- epilogue: bias + ReLU + store; snapshot N==0 column branch-free ----
    float keep0[8];                            // nt==0 values (column N==0 lives here)
#pragma unroll
    for (int nt = 0; nt < 4; ++nt) {
#pragma unroll
        for (int r = 0; r < 8; ++r) {
            const int row = r + hi * 8;        // D: VGPR r -> M = r (+8 for hi half)
            float v = c[nt][r] + bv[nt];
            v = v > 0.f ? v : 0.f;
            y[(size_t)(e0 + row) * OUT_CH + nt * 16 + m] = v;
            if (nt == 0) keep0[r] = v;
        }
    }
    if (m == 0) {                              // single EXEC toggle, 8 stores
#pragma unroll
        for (int r = 0; r < 8; ++r)
            col0[e0 + r + hi * 8] = keep0[r];
    }
}

// ---------------------------------------------------------------------------
// Pass 3: out[src[e], 0] += y_original[e, 0]  (reads the snapshot, preserving
// JAX .at[].add semantics of adding pre-scatter values)
// ---------------------------------------------------------------------------
__global__ __launch_bounds__(256) void scatter_col0(
    const long long* __restrict__ ei,
    const float* __restrict__ col0,
    float* __restrict__ out)
{
    int e = blockIdx.x * blockDim.x + threadIdx.x;
    if (e < N_EDGES) {
        atomicAdd(out + (size_t)((int)ei[e]) * OUT_CH, col0[e]);
    }
}

extern "C" void kernel_launch(void* const* d_in, const int* in_sizes, int n_in,
                              void* d_out, int out_size, void* d_ws, size_t ws_size,
                              hipStream_t stream) {
    const float*     x  = (const float*)d_in[0];
    const long long* ei = (const long long*)d_in[1];   // int64 edge_index
    const float*     W  = (const float*)d_in[2];
    const float*     b  = (const float*)d_in[3];
    float*           out = (float*)d_out;

    // workspace layout (all 16B aligned): xh | wh | col0  (~9.6 MB total)
    char* ws = (char*)d_ws;
    _Float16* xh   = (_Float16*)ws;                                   // 6,400,000 B
    _Float16* wh   = (_Float16*)(ws + (size_t)N_NODES * FEAT * 2);    //    16,384 B
    float*    col0 = (float*)(ws + (size_t)N_NODES * FEAT * 2
                                 + (size_t)OUT_CH * IN_CH * 2);       // 3,200,000 B

    const int nx4 = (N_NODES * FEAT) / 4;              // 800,000 float4 groups
    const int nw4 = (OUT_CH * IN_CH) / 4;              // 2,048
    cvt_f32_f16_x4<<<(nx4 + 255) / 256, 256, 0, stream>>>(x, xh, nx4);
    cvt_f32_f16_x4<<<(nw4 + 255) / 256, 256, 0, stream>>>(W, wh, nw4);

    const int ntiles = N_EDGES / 16;                   // 50,000 tiles, 8 waves/block
    edge_gemm<<<(ntiles + 7) / 8, 256, 0, stream>>>(xh, wh, ei, b, out, col0);

    scatter_col0<<<(N_EDGES + 255) / 256, 256, 0, stream>>>(ei, col0, out);
}
